// TokenCentricGraphAttention_85358180041394
// MI455X (gfx1250) — matrix operand order repaired and checked by reading, not compiled, not run on hardware
//
#include <hip/hip_runtime.h>
#include <hip/hip_bf16.h>
#include <math.h>

// ---------------------------------------------------------------------------
// TokenCentricGraphAttention for MI455X (gfx1250, wave32, WMMA).
//
// Key restructuring (see analysis): only B*E = 8000 gathered rows ever matter
// for Q (src) and K/V (dst); the final @Wo GEMM is linear so it is applied
// per-edge and scatter-added onto an output pre-initialized with bo.
// All big GEMMs use V_WMMA_F32_16X16X32_F16 (f16 operands, f32 accumulate).
// ---------------------------------------------------------------------------

#define BB   4
#define SS   8192
#define DD   1024
#define HH   16
#define HD   64

typedef __attribute__((ext_vector_type(16))) _Float16 v16h;
typedef __attribute__((ext_vector_type(8)))  _Float16 v8h;
typedef __attribute__((ext_vector_type(8)))  float    v8f;

__device__ __forceinline__ v8f wmma_f16(v16h a, v16h b, v8f c) {
    return __builtin_amdgcn_wmma_f32_16x16x32_f16(
        /*neg_a=*/false, a, /*neg_b=*/false, b,
        /*c_mod=*/(short)0, c, /*reuse_a=*/false, /*reuse_b=*/false);
}

__device__ __forceinline__ float gelu_exact(float v) {
    return 0.5f * v * (1.0f + erff(v * 0.70710678118654752f));
}

// B fragment (32x16, KxN) from a pre-transposed weight BT[N][K] (row-major).
// ISA layout: lane c (0-15) holds column n0+c, K = k0..k0+7 and k0+16..k0+23;
// lane c+16 holds K = k0+8..k0+15 and k0+24..k0+31.
__device__ __forceinline__ v16h load_frag_b(const _Float16* __restrict__ BT,
                                            int ldb, int n0, int k0, int lane) {
    int c  = lane & 15;
    int hi = (lane >> 4) & 1;
    const _Float16* p = BT + (size_t)(n0 + c) * ldb + k0 + hi * 8;
    v8h c1 = *(const v8h*)p;
    v8h c2 = *(const v8h*)(p + 16);
    v16h b;
    for (int t = 0; t < 8; ++t) { b[t] = c1[t]; b[8 + t] = c2[t]; }
    return b;
}

// ---------------------------------------------------------------------------
// LDS-tiled f32 -> f16 transpose: WT[n][k] = (f16) W[k][n].  block (32,8).
// ---------------------------------------------------------------------------
__global__ void __launch_bounds__(256)
transpose_f32_to_f16(const float* __restrict__ W, _Float16* __restrict__ WT,
                     int K, int N) {
    __shared__ float tile[32][33];
    int kb = blockIdx.x * 32, nb = blockIdx.y * 32;
    for (int i = threadIdx.y; i < 32; i += 8)
        tile[i][threadIdx.x] = W[(size_t)(kb + i) * N + nb + threadIdx.x];
    __syncthreads();
    for (int i = threadIdx.y; i < 32; i += 8)
        WT[(size_t)(nb + i) * K + kb + threadIdx.x] =
            (_Float16)tile[threadIdx.x][i];
}

// ---------------------------------------------------------------------------
// Gather x rows at src/dst edge indices and convert to f16.
// xs/xd are [B*E, D] row-major f16.
// ---------------------------------------------------------------------------
__global__ void __launch_bounds__(256)
gather_convert(const float* __restrict__ x, const int* __restrict__ src,
               const int* __restrict__ dst, _Float16* __restrict__ xs,
               _Float16* __restrict__ xd, int E) {
    int idx = blockIdx.x * 256 + threadIdx.x;
    int total = BB * E * DD;
    if (idx >= total) return;
    int d   = idx & (DD - 1);
    int row = idx >> 10;          // b*E + e
    int b   = row / E;
    int e   = row - b * E;
    xs[idx] = (_Float16)x[((size_t)b * SS + src[e]) * DD + d];
    xd[idx] = (_Float16)x[((size_t)b * SS + dst[e]) * DD + d];
}

// ---------------------------------------------------------------------------
// Generic WMMA GEMM: C[M,N] = A[M,K] @ B[K,N] (+bias), A f16 row-major,
// B given transposed (BT[N][K] f16).  One wave computes a 64x32 tile
// (4x2 grid of 16x16 f32 accumulators).  Block = 8 waves spanning 256 cols.
// ---------------------------------------------------------------------------
template <typename TO>
__global__ void __launch_bounds__(256)
gemm_wmma(const _Float16* __restrict__ A, const _Float16* __restrict__ BT,
          const float* __restrict__ bias, TO* __restrict__ C,
          int M, int N, int K) {
    int wave = threadIdx.x >> 5;
    int lane = threadIdx.x & 31;
    int m0 = blockIdx.x * 64;
    int n0 = (blockIdx.y * 8 + wave) * 32;
    if (n0 >= N || m0 >= M) return;
    int r  = lane & 15;
    int hi = (lane >> 4) & 1;

    v8f acc[4][2] = {};
    for (int k0 = 0; k0 < K; k0 += 32) {
        v16h bf0 = load_frag_b(BT, K, n0,      k0, lane);
        v16h bf1 = load_frag_b(BT, K, n0 + 16, k0, lane);
#pragma unroll
        for (int i = 0; i < 4; ++i) {
            int row = m0 + i * 16 + r;
            if (row >= M) row = M - 1;              // clamp (guarded on store)
            const _Float16* p = A + (size_t)row * K + k0 + hi * 8;
            v8h c1 = *(const v8h*)p;                // K = k0+hi*8   .. +7
            v8h c2 = *(const v8h*)(p + 16);         // K = k0+16+hi*8.. +7
            v16h a;
#pragma unroll
            for (int t = 0; t < 8; ++t) { a[t] = c1[t]; a[8 + t] = c2[t]; }
            acc[i][0] = wmma_f16(a, bf0, acc[i][0]);
            acc[i][1] = wmma_f16(a, bf1, acc[i][1]);
        }
    }

    int cc = lane & 15;
    float bv0 = bias ? bias[n0 + cc]      : 0.0f;
    float bv1 = bias ? bias[n0 + 16 + cc] : 0.0f;
#pragma unroll
    for (int i = 0; i < 4; ++i)
#pragma unroll
        for (int j = 0; j < 8; ++j) {
            int row = m0 + i * 16 + hi * 8 + j;
            if (row < M) {
                C[(size_t)row * N + n0 + cc]      = (TO)(acc[i][0][j] + bv0);
                C[(size_t)row * N + n0 + 16 + cc] = (TO)(acc[i][1][j] + bv1);
            }
        }
}

// ---------------------------------------------------------------------------
// Edge MLP, fully fused: rows are (b,e,h) triples (row = (b*E+e)*16 + h),
// A row = concat(q_src[b,e,h,:], k_dst[b,e,h,:]) gathered on the fly (K=128),
// B = W1T (64x128).  Epilogue applies bias+GELU, dots with W2, adds b2,
// scales, and reduces across the 16 lanes of each C row with shfl_xor.
// One wave owns a full 64x64 tile (4x4 accumulators) so each row's 64
// hidden values are reducible without LDS.
// ---------------------------------------------------------------------------
__global__ void __launch_bounds__(256)
edge_mlp(const _Float16* __restrict__ qs, const _Float16* __restrict__ kd,
         const _Float16* __restrict__ W1T, const float* __restrict__ b1,
         const float* __restrict__ W2, const float* __restrict__ b2,
         float* __restrict__ scores, int E, int M2) {
    int wave = threadIdx.x >> 5;
    int lane = threadIdx.x & 31;
    int m0 = (blockIdx.x * 8 + wave) * 64;
    if (m0 >= M2) return;
    int r  = lane & 15;
    int hi = (lane >> 4) & 1;

    v8f acc[4][4] = {};
#pragma unroll
    for (int k0 = 0; k0 < 128; k0 += 32) {
        v16h bf[4];
#pragma unroll
        for (int t = 0; t < 4; ++t) bf[t] = load_frag_b(W1T, 128, t * 16, k0, lane);
#pragma unroll
        for (int i = 0; i < 4; ++i) {
            int be = (m0 >> 4) + i;                 // b*E + e  (rows 16-aligned)
            const _Float16* qrow = qs + (size_t)be * DD + r * HD;
            const _Float16* krow = kd + (size_t)be * DD + r * HD;
            int ka = k0 + hi * 8;                   // 8-half chunks never
            int kb = k0 + 16 + hi * 8;              // straddle the 64 boundary
            v8h c1 = (ka < HD) ? *(const v8h*)(qrow + ka) : *(const v8h*)(krow + ka - HD);
            v8h c2 = (kb < HD) ? *(const v8h*)(qrow + kb) : *(const v8h*)(krow + kb - HD);
            v16h a;
#pragma unroll
            for (int t = 0; t < 8; ++t) { a[t] = c1[t]; a[8 + t] = c2[t]; }
#pragma unroll
            for (int t = 0; t < 4; ++t) acc[i][t] = wmma_f16(a, bf[t], acc[i][t]);
        }
    }

    int c = lane & 15;
    float w2v[4], b1v[4];
#pragma unroll
    for (int t = 0; t < 4; ++t) { w2v[t] = W2[t * 16 + c]; b1v[t] = b1[t * 16 + c]; }
    float bb2   = b2[0];
    float scale = 0.125f;                            // HD^-0.5
#pragma unroll
    for (int i = 0; i < 4; ++i) {
#pragma unroll
        for (int j = 0; j < 8; ++j) {
            float s = 0.0f;
#pragma unroll
            for (int t = 0; t < 4; ++t)
                s += gelu_exact(acc[i][t][j] + b1v[t]) * w2v[t];
            // reduce over the 16 lanes holding this C row (xor masks < 16
            // stay inside each half of the wave)
            s += __shfl_xor(s, 1, 32);
            s += __shfl_xor(s, 2, 32);
            s += __shfl_xor(s, 4, 32);
            s += __shfl_xor(s, 8, 32);
            if (c == 0) {
                int row = m0 + i * 16 + hi * 8 + j;  // (b*E+e)*16 + h
                int h   = row & 15;
                int be  = row >> 4;
                int b   = be / E;
                int e   = be - b * E;
                scores[((size_t)b * HH + h) * E + e] = scale * (s + bb2);
            }
        }
    }
}

// ---------------------------------------------------------------------------
// Softmax over all E edges per (b,h), folding in edge_weight[h].
// ---------------------------------------------------------------------------
__global__ void __launch_bounds__(256)
softmax_edges(const float* __restrict__ scores, const float* __restrict__ ew,
              float* __restrict__ attn, int E) {
    int bh = blockIdx.x;                 // 0 .. B*H-1
    const float* s = scores + (size_t)bh * E;
    __shared__ float red[256];
    int tid = threadIdx.x;

    float m = -3.4e38f;
    for (int e = tid; e < E; e += 256) m = fmaxf(m, s[e]);
    red[tid] = m; __syncthreads();
    for (int w = 128; w > 0; w >>= 1) {
        if (tid < w) red[tid] = fmaxf(red[tid], red[tid + w]);
        __syncthreads();
    }
    m = red[0]; __syncthreads();

    float sum = 0.0f;
    for (int e = tid; e < E; e += 256) sum += __expf(s[e] - m);
    red[tid] = sum; __syncthreads();
    for (int w = 128; w > 0; w >>= 1) {
        if (tid < w) red[tid] += red[tid + w];
        __syncthreads();
    }
    sum = red[0];

    float g = ew[bh & (HH - 1)] / sum;   // bh = b*H + h
    for (int e = tid; e < E; e += 256)
        attn[(size_t)bh * E + e] = __expf(s[e] - m) * g;
}

// ---------------------------------------------------------------------------
// weighted[b,e,:] = attn[b, h(d), e] * v_dst[b,e,:]  (f16 for the Wo GEMM)
// ---------------------------------------------------------------------------
__global__ void __launch_bounds__(256)
apply_attn(const float* __restrict__ attn, const _Float16* __restrict__ vd,
           _Float16* __restrict__ we, int E) {
    int idx = blockIdx.x * 256 + threadIdx.x;
    int total = BB * E * DD;
    if (idx >= total) return;
    int d   = idx & (DD - 1);
    int row = idx >> 10;                 // b*E + e
    int h   = d >> 6;
    int b   = row / E;
    int e   = row - b * E;
    float a = attn[((size_t)b * HH + h) * E + e];
    we[idx] = (_Float16)(a * (float)vd[idx]);
}

// out[b,s,:] = bo  (everything not touched by an edge stays at bo)
__global__ void __launch_bounds__(256)
init_out(float* __restrict__ out, const float* __restrict__ bo) {
    int idx = blockIdx.x * 256 + threadIdx.x;
    out[idx] = bo[idx & (DD - 1)];
}

// out[b, src[e], :] += ze[b*E+e, :]   (duplicate src indices -> atomics)
__global__ void __launch_bounds__(256)
scatter_edges(const float* __restrict__ ze, const int* __restrict__ src,
              float* __restrict__ out, int E) {
    int idx = blockIdx.x * 256 + threadIdx.x;
    int total = BB * E * DD;
    if (idx >= total) return;
    int d   = idx & (DD - 1);
    int row = idx >> 10;                 // b*E + e
    int b   = row / E;
    int e   = row - b * E;
    int s   = src[e];
    atomicAdd(out + ((size_t)b * SS + s) * DD + d, ze[idx]);
}

// ---------------------------------------------------------------------------
extern "C" void kernel_launch(void* const* d_in, const int* in_sizes, int n_in,
                              void* d_out, int out_size, void* d_ws, size_t ws_size,
                              hipStream_t stream) {
    const float* x   = (const float*)d_in[0];
    const int*   src = (const int*)  d_in[1];
    const int*   dst = (const int*)  d_in[2];
    const float* Wq  = (const float*)d_in[3];
    const float* bq  = (const float*)d_in[4];
    const float* Wk  = (const float*)d_in[5];
    const float* bk  = (const float*)d_in[6];
    const float* Wv  = (const float*)d_in[7];
    const float* bv  = (const float*)d_in[8];
    const float* Wo  = (const float*)d_in[9];
    const float* bo  = (const float*)d_in[10];
    const float* ew  = (const float*)d_in[11];
    const float* W1  = (const float*)d_in[12];
    const float* b1  = (const float*)d_in[13];
    const float* W2  = (const float*)d_in[14];
    const float* b2  = (const float*)d_in[15];
    float* out = (float*)d_out;

    const int E  = in_sizes[1];
    const int M  = BB * E;          // gathered rows
    const int M2 = M * HH;          // edge-head rows

    // ---- carve workspace (256B aligned regions) ----
    char* p = (char*)d_ws;
    auto carve = [&](size_t bytes) -> char* {
        char* r = p; p += (bytes + 255) & ~(size_t)255; return r;
    };
    _Float16* WqT = (_Float16*)carve((size_t)DD * DD * 2);
    _Float16* WkT = (_Float16*)carve((size_t)DD * DD * 2);
    _Float16* WvT = (_Float16*)carve((size_t)DD * DD * 2);
    _Float16* WoT = (_Float16*)carve((size_t)DD * DD * 2);
    _Float16* W1T = (_Float16*)carve((size_t)128 * 64 * 2);
    _Float16* xsh = (_Float16*)carve((size_t)M * DD * 2);
    _Float16* xdh = (_Float16*)carve((size_t)M * DD * 2);
    _Float16* qsh = (_Float16*)carve((size_t)M * DD * 2);
    _Float16* kdh = (_Float16*)carve((size_t)M * DD * 2);
    _Float16* vdh = (_Float16*)carve((size_t)M * DD * 2);
    float*    sco = (float*)   carve((size_t)BB * HH * E * 4);
    float*    att = (float*)   carve((size_t)BB * HH * E * 4);
    _Float16* weh = (_Float16*)carve((size_t)M * DD * 2);
    float*    ze  = (float*)   carve((size_t)M * DD * 4);

    dim3 tb(32, 8);
    // ---- weight prep ----
    transpose_f32_to_f16<<<dim3(DD / 32, DD / 32), tb, 0, stream>>>(Wq, WqT, DD, DD);
    transpose_f32_to_f16<<<dim3(DD / 32, DD / 32), tb, 0, stream>>>(Wk, WkT, DD, DD);
    transpose_f32_to_f16<<<dim3(DD / 32, DD / 32), tb, 0, stream>>>(Wv, WvT, DD, DD);
    transpose_f32_to_f16<<<dim3(DD / 32, DD / 32), tb, 0, stream>>>(Wo, WoT, DD, DD);
    transpose_f32_to_f16<<<dim3(128 / 32, 64 / 32), tb, 0, stream>>>(W1, W1T, 128, 64);

    // ---- gather + convert only the rows that matter ----
    int nGE = M * DD;
    gather_convert<<<(nGE + 255) / 256, 256, 0, stream>>>(x, src, dst, xsh, xdh, E);

    // ---- Q/K/V GEMMs on gathered rows (f16 WMMA, f32 accum, f16 out) ----
    dim3 gg((M + 63) / 64, DD / 256);
    gemm_wmma<_Float16><<<gg, 256, 0, stream>>>(xsh, WqT, bq, qsh, M, DD, DD);
    gemm_wmma<_Float16><<<gg, 256, 0, stream>>>(xdh, WkT, bk, kdh, M, DD, DD);
    gemm_wmma<_Float16><<<gg, 256, 0, stream>>>(xdh, WvT, bv, vdh, M, DD, DD);

    // ---- fused edge MLP (GEMM + bias + GELU + W2 dot + scale) ----
    edge_mlp<<<(M2 + 511) / 512, 256, 0, stream>>>(qsh, kdh, W1T, b1, W2, b2,
                                                   sco, E, M2);

    // ---- softmax over all edges per (b,h), * edge_weight ----
    softmax_edges<<<BB * HH, 256, 0, stream>>>(sco, ew, att, E);

    // ---- weighted values, per-edge Wo GEMM ----
    apply_attn<<<(nGE + 255) / 256, 256, 0, stream>>>(att, vdh, weh, E);
    gemm_wmma<float><<<gg, 256, 0, stream>>>(weh, WoT, nullptr, ze, M, DD, DD);

    // ---- output: bo everywhere, + scatter-add of per-edge contributions ----
    init_out<<<(BB * SS * DD) / 256, 256, 0, stream>>>(out, bo);
    scatter_edges<<<(nGE + 255) / 256, 256, 0, stream>>>(ze, src, out, E);
}
